// TopologyEncoder_39642548142641
// MI455X (gfx1250) — matrix operand order repaired and checked by reading, not compiled, not run on hardware
//
#include <hip/hip_runtime.h>
#include <hip/hip_bf16.h>

#define NNODES 50000
#define NEDGES 800000
#define DINF   16
#define HDIM   128
#define NHEADS 8
#define CDIM   16
#define NLEV   4
#define ETOT   (NEDGES + NNODES)
#define EPSV   1e-5f

typedef __attribute__((ext_vector_type(16))) _Float16 v16h;
typedef __attribute__((ext_vector_type(8)))  float    v8f;

// ---------- helpers ----------
__device__ __forceinline__ unsigned fflip(float f) {
  unsigned u = __float_as_uint(f);
  return (u >> 31) ? ~u : (u | 0x80000000u);
}
__device__ __forceinline__ float funflip(unsigned m) {
  unsigned u = (m >> 31) ? (m & 0x7FFFFFFFu) : ~m;
  return __uint_as_float(u);
}
__device__ __forceinline__ float gelu_exact(float v) {
  return 0.5f * v * (1.0f + erff(v * 0.70710678118654752440f));
}

// ---------- zero-pad [N,16] -> [N,32] so GEMM K is always a multiple of 32 ----------
__global__ void pad_a_kernel(const float* __restrict__ x, float* __restrict__ out, int N) {
  int i = blockIdx.x * 256 + threadIdx.x;
  if (i >= N * 32) return;
  int n = i >> 5, c = i & 31;
  out[i] = (c < DINF) ? x[(size_t)n * DINF + c] : 0.0f;
}

// ---------- weight packing into WMMA B-fragment layout ----------
// packed[((kt*Ntiles + nt)*32 + lane)*16 + j]  <-  W[k*N + n]
//   k = kt*32 + (lane/16)*16 + j ,  n = nt*16 + (lane%16)
__global__ void pack_w_kernel(const float* __restrict__ W, _Float16* __restrict__ P,
                              int K, int Nn) {
  int Ntiles = Nn >> 4;
  int Ktiles = (K + 31) >> 5;
  int total = Ktiles * Ntiles * 512;
  int idx = blockIdx.x * 256 + threadIdx.x;
  if (idx >= total) return;
  int j    = idx & 15;
  int lane = (idx >> 4) & 31;
  int tile = idx >> 9;
  int nt = tile % Ntiles;
  int kt = tile / Ntiles;
  int k = (kt << 5) + ((lane >> 4) << 4) + j;
  int n = (nt << 4) + (lane & 15);
  float v = (k < K) ? W[(size_t)k * Nn + n] : 0.0f;
  P[idx] = (_Float16)v;
}

// ---------- WMMA GEMM: C[M,Nn] = A[M,K](f32->f16) @ Bpacked + bias, epilogue ----------
// Requirements: M % 16 == 0, K % 32 == 0, Nn % 64 == 0.
// Each wave: one 16-row M-tile x four 16-col N-tiles (4 accumulators).
// mode: 0 = bias, 1 = bias+GELU(exact), 2 = bias+ReLU
__global__ void gemm_wmma_kernel(const float* __restrict__ A, const _Float16* __restrict__ Bp,
                                 const float* __restrict__ bias, float* __restrict__ Cout,
                                 int M, int Nn, int K, int lda, int mode) {
  int wave = threadIdx.x >> 5;
  int lane = threadIdx.x & 31;
  int Mtiles = M >> 4;
  int mtile = blockIdx.x * 8 + wave;
  if (mtile >= Mtiles) return;           // uniform per wave: EXEC stays all-ones
  int Ntiles = Nn >> 4;
  int nt0 = blockIdx.y << 2;             // 4 N-tiles per wave
  int Ktiles = K >> 5;
  int sel = lane >> 4;                   // half-wave select
  int m = (mtile << 4) + (lane & 15);
  const float* Arow = A + (size_t)m * lda;

  v8f acc[4];
#pragma unroll
  for (int t = 0; t < 4; ++t)
#pragma unroll
    for (int r = 0; r < 8; ++r) acc[t][r] = 0.0f;

  for (int kt = 0; kt < Ktiles; ++kt) {
    int k0 = kt << 5;
    // A fragment: lanes 0-15 -> M=lane, k runs [k0+sel*16, +16)
    v8f fa = *reinterpret_cast<const v8f*>(Arow + k0 + sel * 16);
    v8f fb = *reinterpret_cast<const v8f*>(Arow + k0 + sel * 16 + 8);
    v16h a;
#pragma unroll
    for (int j = 0; j < 8; ++j) {
      a[j]     = (_Float16)fa[j];
      a[8 + j] = (_Float16)fb[j];
    }
    const _Float16* brow = Bp + (((size_t)kt * Ntiles + nt0) * 32 + lane) * 16;
#pragma unroll
    for (int t = 0; t < 4; ++t) {
      v16h b = *reinterpret_cast<const v16h*>(brow + (size_t)t * 512);
      acc[t] = __builtin_amdgcn_wmma_f32_16x16x32_f16(false, a, false, b, (short)0, acc[t],
                                                      false, false);
    }
  }

  int rbase = (mtile << 4) + sel * 8;    // D layout: lanes 0-15 -> M=r, lanes 16-31 -> M=8+r
#pragma unroll
  for (int t = 0; t < 4; ++t) {
    int col = ((nt0 + t) << 4) + (lane & 15);
    float bv = bias ? bias[col] : 0.0f;
#pragma unroll
    for (int r = 0; r < 8; ++r) {
      float v = acc[t][r] + bv;
      if (mode == 1) v = gelu_exact(v);
      else if (mode == 2) v = v > 0.0f ? v : 0.0f;
      Cout[(size_t)(rbase + r) * Nn + col] = v;
    }
  }
}

// ---------- GATv2 edge kernels ----------
__global__ void edge_score_kernel(const int* __restrict__ src, const int* __restrict__ dst,
                                  const float* __restrict__ xl, const float* __restrict__ xr,
                                  const float* __restrict__ att,
                                  float* __restrict__ score, unsigned* __restrict__ smax) {
  int tid = blockIdx.x * 256 + threadIdx.x;
  if (tid >= ETOT * NHEADS) return;
  int e = tid >> 3, hd = tid & 7;
  int s = (e < NEDGES) ? src[e] : (e - NEDGES);
  int d = (e < NEDGES) ? dst[e] : (e - NEDGES);
  const float* pl = xl + (size_t)s * HDIM + hd * CDIM;
  const float* pr = xr + (size_t)d * HDIM + hd * CDIM;
  const float* pa = att + hd * CDIM;
  float sc = 0.0f;
#pragma unroll
  for (int c = 0; c < CDIM; ++c) {
    float v = pl[c] + pr[c];
    v = v > 0.0f ? v : 0.2f * v;
    sc += pa[c] * v;
  }
  score[tid] = sc;
  atomicMax(smax + (size_t)d * NHEADS + hd, fflip(sc));
}

__global__ void edge_exp_kernel(const int* __restrict__ dst,
                                const unsigned* __restrict__ smax,
                                float* __restrict__ score, float* __restrict__ denom) {
  int tid = blockIdx.x * 256 + threadIdx.x;
  if (tid >= ETOT * NHEADS) return;
  int e = tid >> 3, hd = tid & 7;
  int d = (e < NEDGES) ? dst[e] : (e - NEDGES);
  float mx = funflip(smax[(size_t)d * NHEADS + hd]);
  float ex = expf(score[tid] - mx);
  score[tid] = ex;
  atomicAdd(denom + (size_t)d * NHEADS + hd, ex);
}

__global__ void edge_agg_kernel(const int* __restrict__ src, const int* __restrict__ dst,
                                const float* __restrict__ score, const float* __restrict__ denom,
                                const float* __restrict__ xl, float* __restrict__ agg) {
  int tid = blockIdx.x * 256 + threadIdx.x;
  if (tid >= ETOT * NHEADS) return;
  int e = tid >> 3, hd = tid & 7;
  int s = (e < NEDGES) ? src[e] : (e - NEDGES);
  int d = (e < NEDGES) ? dst[e] : (e - NEDGES);
  float alpha = score[tid] / denom[(size_t)d * NHEADS + hd];
  const float* pl = xl + (size_t)s * HDIM + hd * CDIM;
  float* pd = agg + (size_t)d * HDIM + hd * CDIM;
#pragma unroll
  for (int c = 0; c < CDIM; ++c) atomicAdd(pd + c, alpha * pl[c]);
}

// ---------- elementwise ----------
__global__ void add3_kernel(float* __restrict__ out, const float* __restrict__ a,
                            const float* __restrict__ b, const float* __restrict__ rowv,
                            int total) {
  int i = blockIdx.x * 256 + threadIdx.x;
  if (i >= total) return;
  float v = a[i] + b[i];
  if (rowv) v += rowv[i & (HDIM - 1)];
  out[i] = v;
}

__global__ void add_rowvec_kernel(float* __restrict__ out, const float* __restrict__ a,
                                  const float* __restrict__ rowv, int total) {
  int i = blockIdx.x * 256 + threadIdx.x;
  if (i >= total) return;
  out[i] = a[i] + rowv[i & (HDIM - 1)];
}

// ---------- GraphNorm (column stats over N, H=128) ----------
__global__ void colsum_kernel(const float* __restrict__ x, float* __restrict__ sums, int N) {
  __shared__ float sm[256];
  int t = threadIdx.x;
  int col = t & 127, rp = t >> 7;
  int r0 = blockIdx.x * 256;
  int rend = r0 + 256; if (rend > N) rend = N;
  float local = 0.0f;
  for (int r = r0 + rp; r < rend; r += 2) local += x[(size_t)r * HDIM + col];
  sm[t] = local; __syncthreads();
  if (t < 128) atomicAdd(&sums[col], sm[t] + sm[t + 128]);
}

__global__ void colvar_kernel(const float* __restrict__ x, const float* __restrict__ sums,
                              float* __restrict__ varsums, const float* __restrict__ ms,
                              int N, float invN) {
  __shared__ float sm[256];
  int t = threadIdx.x;
  int col = t & 127, rp = t >> 7;
  int r0 = blockIdx.x * 256;
  int rend = r0 + 256; if (rend > N) rend = N;
  float mu = sums[col] * invN;
  float msv = ms[col];
  float local = 0.0f;
  for (int r = r0 + rp; r < rend; r += 2) {
    float d = x[(size_t)r * HDIM + col] - msv * mu;
    local += d * d;
  }
  sm[t] = local; __syncthreads();
  if (t < 128) atomicAdd(&varsums[col], sm[t] + sm[t + 128]);
}

__global__ void gn_norm_kernel(const float* __restrict__ x, const float* __restrict__ sums,
                               const float* __restrict__ varsums, const float* __restrict__ w,
                               const float* __restrict__ b, const float* __restrict__ ms,
                               float* __restrict__ out, int total, float invN) {
  int i = blockIdx.x * 256 + threadIdx.x;
  if (i >= total) return;
  int col = i & 127;
  float mu = sums[col] * invN;
  float var = varsums[col] * invN;
  out[i] = w[col] * (x[i] - ms[col] * mu) * rsqrtf(var + EPSV) + b[col];
}

// ---------- final-stage kernels ----------
__global__ void concat_kernel(const float* __restrict__ outs, float* __restrict__ big, int total) {
  int i = blockIdx.x * 256 + threadIdx.x;
  if (i >= total) return;
  int n = i >> 9, r = i & 511;
  int h = r >> 2, l = r & 3;
  big[i] = outs[(size_t)l * NNODES * HDIM + (size_t)n * HDIM + h];
}

__global__ void attn2_softmax_kernel(const float* __restrict__ relu_out,
                                     const float* __restrict__ W2, const float* __restrict__ b2,
                                     float* __restrict__ wout, int N) {
  int n = blockIdx.x * 256 + threadIdx.x;
  if (n >= N) return;
  const float* row = relu_out + (size_t)n * HDIM;
  float sc[NLEV];
#pragma unroll
  for (int l = 0; l < NLEV; ++l) sc[l] = b2[l];
  for (int h = 0; h < HDIM; ++h) {
    float rv = row[h];
#pragma unroll
    for (int l = 0; l < NLEV; ++l) sc[l] += rv * W2[h * NLEV + l];
  }
  float mx = sc[0];
#pragma unroll
  for (int l = 1; l < NLEV; ++l) mx = sc[l] > mx ? sc[l] : mx;
  float se = 0.0f;
#pragma unroll
  for (int l = 0; l < NLEV; ++l) { sc[l] = expf(sc[l] - mx); se += sc[l]; }
  float inv = 1.0f / se;
#pragma unroll
  for (int l = 0; l < NLEV; ++l) wout[(size_t)n * NLEV + l] = sc[l] * inv;
}

__global__ void combine_kernel(const float* __restrict__ outs, const float* __restrict__ wv,
                               float* __restrict__ out, int total) {
  int i = blockIdx.x * 256 + threadIdx.x;
  if (i >= total) return;
  int n = i >> 7;
  float acc = 0.0f;
#pragma unroll
  for (int l = 0; l < NLEV; ++l)
    acc += outs[(size_t)l * NNODES * HDIM + i] * wv[(size_t)n * NLEV + l];
  out[i] = acc;
}

__global__ void ln_gelu_kernel(const float* __restrict__ x, const float* __restrict__ w,
                               const float* __restrict__ b, float* __restrict__ out, int N) {
  __shared__ float red[HDIM];
  int n = blockIdx.x, t = threadIdx.x;
  float v = x[(size_t)n * HDIM + t];
  red[t] = v; __syncthreads();
  for (int s = 64; s > 0; s >>= 1) { if (t < s) red[t] += red[t + s]; __syncthreads(); }
  float mu = red[0] * (1.0f / HDIM);
  __syncthreads();
  float c = v - mu;
  red[t] = c * c; __syncthreads();
  for (int s = 64; s > 0; s >>= 1) { if (t < s) red[t] += red[t + s]; __syncthreads(); }
  float var = red[0] * (1.0f / HDIM);
  float y = w[t] * c * rsqrtf(var + EPSV) + b[t];
  out[(size_t)n * HDIM + t] = gelu_exact(y);
}

__global__ void scale_store_kernel(const float* __restrict__ sums, float* __restrict__ out,
                                   float scale) {
  int t = threadIdx.x;
  if (t < HDIM) out[t] = sums[t] * scale;
}

// ---------- host orchestration ----------
extern "C" void kernel_launch(void* const* d_in, const int* in_sizes, int n_in,
                              void* d_out, int out_size, void* d_ws, size_t ws_size,
                              hipStream_t stream) {
  (void)in_sizes; (void)n_in; (void)out_size; (void)ws_size;

  const float* node_features = (const float*)d_in[0];
  const int*   edge_index    = (const int*)d_in[1];
  const int*   src = edge_index;
  const int*   dst = edge_index + NEDGES;
  const float* inproj0_W = (const float*)d_in[2];
  const float* inproj0_b = (const float*)d_in[3];
  const float* inproj_W  = (const float*)d_in[4];
  const float* inproj_b  = (const float*)d_in[5];
  const float* gat_Wl    = (const float*)d_in[6];
  const float* gat_Wr    = (const float*)d_in[7];
  const float* gat_att   = (const float*)d_in[8];
  const float* gat_b     = (const float*)d_in[9];
  const float* norm_w    = (const float*)d_in[10];
  const float* norm_b    = (const float*)d_in[11];
  const float* norm_ms   = (const float*)d_in[12];
  const float* ffn_W1    = (const float*)d_in[13];
  const float* ffn_b1    = (const float*)d_in[14];
  const float* ffn_W2    = (const float*)d_in[15];
  const float* ffn_b2    = (const float*)d_in[16];
  const float* fnorm_w   = (const float*)d_in[17];
  const float* fnorm_b   = (const float*)d_in[18];
  const float* fnorm_ms  = (const float*)d_in[19];
  const float* outproj_W = (const float*)d_in[20];
  const float* outproj_b = (const float*)d_in[21];
  const float* conn_W    = (const float*)d_in[22];
  const float* conn_b    = (const float*)d_in[23];
  const float* attn_W1   = (const float*)d_in[24];
  const float* attn_b1   = (const float*)d_in[25];
  const float* attn_W2   = (const float*)d_in[26];
  const float* attn_b2   = (const float*)d_in[27];
  const float* fin_W1    = (const float*)d_in[28];
  const float* fin_b1    = (const float*)d_in[29];
  const float* ln_w      = (const float*)d_in[30];
  const float* ln_b      = (const float*)d_in[31];
  const float* fin_W2    = (const float*)d_in[32];
  const float* fin_b2    = (const float*)d_in[33];
  const float* res_emb   = (const float*)d_in[34];

  // workspace carve-out (256B aligned)
  char* p = (char*)d_ws;
  auto carve = [&](size_t bytes) -> void* {
    void* r = (void*)p;
    p += (bytes + 255) & ~(size_t)255;
    return r;
  };
  const size_t NH = (size_t)NNODES * HDIM;
  float*    f_cur   = (float*)carve(NH * 4);
  float*    f_h     = (float*)carve(NH * 4);
  float*    f_xl    = (float*)carve(NH * 4);
  float*    f_xr    = (float*)carve(NH * 4);
  float*    f_agg   = (float*)carve(NH * 4);
  float*    f_big   = (float*)carve((size_t)NNODES * 512 * 4);
  float*    f_outs  = (float*)carve((size_t)NLEV * NH * 4);
  float*    f_score = (float*)carve((size_t)ETOT * NHEADS * 4);
  unsigned* u_smax  = (unsigned*)carve((size_t)NNODES * NHEADS * 4);
  float*    f_denom = (float*)carve((size_t)NNODES * NHEADS * 4);
  float*    f_sums  = (float*)carve(HDIM * 4);
  float*    f_vars  = (float*)carve(HDIM * 4);
  float*    f_w     = (float*)carve((size_t)NNODES * NLEV * 4);
  float*    f_pad   = (float*)carve((size_t)NNODES * 32 * 4);
  _Float16* h_W1    = (_Float16*)carve((size_t)512 * 128 * 2);
  _Float16* h_W2    = (_Float16*)carve((size_t)512 * 128 * 2);

  const int EWTOT = ETOT * NHEADS;
  const int egrid = (EWTOT + 255) / 256;
  const int nhgrid = (int)((NH + 255) / 256);
  const int colgrid = (NNODES + 255) / 256;
  const float invN = 1.0f / (float)NNODES;

  auto pack = [&](const float* W, _Float16* P, int K, int Nn) {
    int total = ((K + 31) / 32) * (Nn / 16) * 512;
    pack_w_kernel<<<(total + 255) / 256, 256, 0, stream>>>(W, P, K, Nn);
  };
  auto gemm = [&](const float* A, const _Float16* Bp, const float* bias, float* C,
                  int Nn, int K, int mode) {
    dim3 g((NNODES / 16 + 7) / 8, Nn / 64);   // 4 N-tiles per wave
    gemm_wmma_kernel<<<g, 256, 0, stream>>>(A, Bp, bias, C, NNODES, Nn, K, K, mode);
  };
  auto graph_norm = [&](const float* x, const float* w, const float* b, const float* ms,
                        float* out) {
    hipMemsetAsync(f_sums, 0, HDIM * 4, stream);
    hipMemsetAsync(f_vars, 0, HDIM * 4, stream);
    colsum_kernel<<<colgrid, 256, 0, stream>>>(x, f_sums, NNODES);
    colvar_kernel<<<colgrid, 256, 0, stream>>>(x, f_sums, f_vars, ms, NNODES, invN);
    gn_norm_kernel<<<nhgrid, 256, 0, stream>>>(x, f_sums, f_vars, w, b, ms, out, (int)NH, invN);
  };

  for (int i = 0; i < NLEV; ++i) {
    // input projection
    if (i == 0) {
      pad_a_kernel<<<(NNODES * 32 + 255) / 256, 256, 0, stream>>>(node_features, f_pad, NNODES);
      pack(inproj0_W, h_W1, DINF, HDIM);   // zero-pads K 16 -> 32
      gemm(f_pad, h_W1, inproj0_b, f_h, HDIM, 32, 0);
    } else {
      add_rowvec_kernel<<<nhgrid, 256, 0, stream>>>(f_xl, f_cur, res_emb + i * HDIM, (int)NH);
      pack(inproj_W + (size_t)(i - 1) * HDIM * HDIM, h_W1, HDIM, HDIM);
      gemm(f_xl, h_W1, inproj_b + (size_t)(i - 1) * HDIM, f_h, HDIM, HDIM, 0);
    }
    // two GATv2 layers
    for (int l = 0; l < 2; ++l) {
      int il = i * 2 + l;
      pack(gat_Wl + (size_t)il * HDIM * HDIM, h_W1, HDIM, HDIM);
      pack(gat_Wr + (size_t)il * HDIM * HDIM, h_W2, HDIM, HDIM);
      gemm(f_h, h_W1, nullptr, f_xl, HDIM, HDIM, 0);
      gemm(f_h, h_W2, nullptr, f_xr, HDIM, HDIM, 0);
      hipMemsetAsync(u_smax, 0, (size_t)NNODES * NHEADS * 4, stream);  // fflip(x) > 0 for reals
      hipMemsetAsync(f_denom, 0, (size_t)NNODES * NHEADS * 4, stream);
      hipMemsetAsync(f_agg, 0, NH * 4, stream);
      const float* att = gat_att + (size_t)il * NHEADS * CDIM;
      edge_score_kernel<<<egrid, 256, 0, stream>>>(src, dst, f_xl, f_xr, att, f_score, u_smax);
      edge_exp_kernel<<<egrid, 256, 0, stream>>>(dst, u_smax, f_score, f_denom);
      edge_agg_kernel<<<egrid, 256, 0, stream>>>(src, dst, f_score, f_denom, f_xl, f_agg);
      // h + (agg + gat_b), then GraphNorm -> f_h
      add3_kernel<<<nhgrid, 256, 0, stream>>>(f_xr, f_h, f_agg, gat_b + (size_t)il * HDIM, (int)NH);
      graph_norm(f_xr, norm_w + (size_t)il * HDIM, norm_b + (size_t)il * HDIM,
                 norm_ms + (size_t)il * HDIM, f_h);
    }
    // FFN
    pack(ffn_W1 + (size_t)i * HDIM * 512, h_W1, HDIM, 512);
    gemm(f_h, h_W1, ffn_b1 + (size_t)i * 512, f_big, 512, HDIM, 1 /*gelu*/);
    pack(ffn_W2 + (size_t)i * 512 * HDIM, h_W2, 512, HDIM);
    gemm(f_big, h_W2, ffn_b2 + (size_t)i * HDIM, f_agg, HDIM, 512, 0);
    add3_kernel<<<nhgrid, 256, 0, stream>>>(f_xr, f_h, f_agg, nullptr, (int)NH);
    graph_norm(f_xr, fnorm_w + (size_t)i * HDIM, fnorm_b + (size_t)i * HDIM,
               fnorm_ms + (size_t)i * HDIM, f_h);
    // output projection
    pack(outproj_W + (size_t)i * HDIM * HDIM, h_W1, HDIM, HDIM);
    gemm(f_h, h_W1, outproj_b + (size_t)i * HDIM, f_outs + (size_t)i * NH, HDIM, HDIM, 0);
    // connection to next level
    if (i < NLEV - 1) {
      pack(conn_W + (size_t)i * HDIM * HDIM, h_W2, HDIM, HDIM);
      gemm(f_outs + (size_t)i * NH, h_W2, conn_b + (size_t)i * HDIM, f_cur, HDIM, HDIM, 0);
    }
  }

  // level-attention combine
  int cgrid = (int)(((size_t)NNODES * 512 + 255) / 256);
  concat_kernel<<<cgrid, 256, 0, stream>>>(f_outs, f_big, NNODES * 512);
  pack(attn_W1, h_W1, 512, HDIM);
  gemm(f_big, h_W1, attn_b1, f_xl, HDIM, 512, 2 /*relu*/);
  attn2_softmax_kernel<<<colgrid, 256, 0, stream>>>(f_xl, attn_W2, attn_b2, f_w, NNODES);
  combine_kernel<<<nhgrid, 256, 0, stream>>>(f_outs, f_w, f_agg, (int)NH);

  // final MLP
  pack(fin_W1, h_W1, HDIM, HDIM);
  gemm(f_agg, h_W1, fin_b1, f_xl, HDIM, HDIM, 0);
  ln_gelu_kernel<<<NNODES, HDIM, 0, stream>>>(f_xl, ln_w, ln_b, f_xr, NNODES);
  pack(fin_W2, h_W2, HDIM, HDIM);
  float* node_emb = (float*)d_out;
  gemm(f_xr, h_W2, fin_b2, node_emb, HDIM, HDIM, 0);

  // graph embedding = column mean of node_emb -> d_out[N*H .. N*H+127]
  hipMemsetAsync(f_sums, 0, HDIM * 4, stream);
  colsum_kernel<<<colgrid, 256, 0, stream>>>(node_emb, f_sums, NNODES);
  scale_store_kernel<<<1, HDIM, 0, stream>>>(f_sums, node_emb + NH, invN);
}
